// ANEStatefulPrefillLayer_7825430413681
// MI455X (gfx1250) — matrix operand order repaired and checked by reading, not compiled, not run on hardware
//
#include <hip/hip_runtime.h>
#include <hip/hip_bf16.h>

// ---------------------------------------------------------------------------
// Problem constants (from the reference)
// ---------------------------------------------------------------------------
constexpr int H    = 16;
constexpr int HKV  = 8;
constexpr int D    = 128;
constexpr int T    = 2048;
constexpr int S    = 4096;
constexpr int HID  = 2048;
constexpr int IDIM = 8192;
constexpr int NREP = H / HKV;
constexpr float EPS   = 1e-6f;
constexpr float SCALE = 0.08838834764831845f;   // 1/sqrt(128)

typedef __attribute__((ext_vector_type(16))) __bf16 v16bf;
typedef __attribute__((ext_vector_type(8)))  float  v8f;
typedef __attribute__((ext_vector_type(4)))  unsigned v4u;
typedef __attribute__((ext_vector_type(8)))  int   v8i_t;
typedef __attribute__((ext_vector_type(4)))  int   v4i_t;

union Frag {
    v16bf v;
    uint4 u[2];
};

// Raw LDS byte offset of a pointer into __shared__ memory (addrspace(3)).
__device__ __forceinline__ unsigned lds_addr(void* p) {
    return (unsigned)(unsigned long long)(__attribute__((address_space(3))) char*)p;
}

// Async global->LDS copy of 16 bytes, tracked by ASYNCcnt.
// LDS dest at byte offset %0 + OFF; global src at %2 (SGPR64 base) + %1 (VGPR32 offset) + OFF
#define ASYNC_B128(ldsOff, vOff, sBase, OFF)                                   \
    asm volatile("global_load_async_to_lds_b128 %0, %1, %2 offset:" #OFF       \
                 :: "v"(ldsOff), "v"(vOff), "s"(sBase) : "memory")

#define WAIT_ASYNC() asm volatile("s_wait_asynccnt 0" ::: "memory")

#if __has_builtin(__builtin_amdgcn_tensor_load_to_lds)
#define HAVE_TDM 1
#else
#define HAVE_TDM 0
#endif

#if HAVE_TDM
// Issue one TDM 2D tile load: tile (tile0 x tile1) elements (2 bytes each)
// from row-major global (row stride = strideElems) into LDS at ldsDst.
// D# layout per CDNA5 ISA ch. 8 (group0: count/lds/global/type; group1:
// data_size, tensor dims, tile dims, dim0 stride). Groups 2/3 unused (<=2D).
__device__ __forceinline__ void tdm_load_tile(
    unsigned ldsDst, const void* gsrc, unsigned dim0, unsigned dim1,
    unsigned tile0, unsigned tile1, unsigned strideElems)
{
    unsigned long long ga = (unsigned long long)gsrc;
    v4u g0;
    g0[0] = 1u;                                            // count = 1
    g0[1] = ldsDst;                                        // lds_addr
    g0[2] = (unsigned)(ga & 0xffffffffull);                // global_addr lo
    g0[3] = (unsigned)((ga >> 32) & 0x1ffffffull) | (2u << 30); // hi | type=2
    v8i_t g1;
    g1[0] = (int)(1u << 16);                               // data_size = 2B
    g1[1] = (int)((dim0 & 0xffffu) << 16);                 // tensor_dim0 lo
    g1[2] = (int)((dim0 >> 16) | ((dim1 & 0xffffu) << 16));// dim0 hi|dim1 lo
    g1[3] = (int)((dim1 >> 16) | (tile0 << 16));           // dim1 hi|tile0
    g1[4] = (int)tile1;                                    // tile_dim1
    g1[5] = (int)strideElems;                              // dim0_stride lo
    g1[6] = 0;
    g1[7] = 0;
    v4i_t z = {};
#if __clang_major__ >= 23
    v8i_t z8 = {};
    __builtin_amdgcn_tensor_load_to_lds(g0, g1, z, z, z8, 0);
#else
    __builtin_amdgcn_tensor_load_to_lds(g0, g1, z, z, 0);
#endif
}
#endif

__device__ __forceinline__ void wait_tensor0() {
#if __has_builtin(__builtin_amdgcn_s_wait_tensorcnt)
    __builtin_amdgcn_s_wait_tensorcnt(0);
#else
    asm volatile("s_wait_tensorcnt 0" ::: "memory");
#endif
}

// ---------------------------------------------------------------------------
// fp32 -> bf16 conversion (weights / activations)
// ---------------------------------------------------------------------------
__global__ __launch_bounds__(256) void f32_to_bf16_kernel(
    const float* __restrict__ in, __bf16* __restrict__ out, long n)
{
    long i = (long)blockIdx.x * blockDim.x + threadIdx.x;
    long stride = (long)gridDim.x * blockDim.x;
    for (; i < n; i += stride) out[i] = (__bf16)in[i];
}

// ---------------------------------------------------------------------------
// RMSNorm: out(bf16, rows x cols row-major) = rms(x) * w
// x[t,c] read at x[t*srow + c*scol]: handles transposed hidden_conv
// (srow=1, scol=T) and row-major hidden (srow=HID, scol=1). cols == 2048.
// ---------------------------------------------------------------------------
__global__ __launch_bounds__(256) void rms_kernel(
    const float* __restrict__ x, const float* __restrict__ w,
    __bf16* __restrict__ out, int cols, long srow, long scol)
{
    const int t = blockIdx.x;
    __shared__ float red[8];
    float loc[8];
    float ss = 0.f;
#pragma unroll
    for (int i = 0; i < 8; ++i) {
        int c = threadIdx.x + i * 256;
        float v = x[(long)t * srow + (long)c * scol];
        loc[i] = v;
        ss += v * v;
    }
#pragma unroll
    for (int off = 16; off >= 1; off >>= 1) ss += __shfl_xor(ss, off, 32);
    if ((threadIdx.x & 31) == 0) red[threadIdx.x >> 5] = ss;
    __syncthreads();
    float tot = 0.f;
#pragma unroll
    for (int i = 0; i < 8; ++i) tot += red[i];
    float inv = rsqrtf(tot / (float)cols + EPS);
#pragma unroll
    for (int i = 0; i < 8; ++i) {
        int c = threadIdx.x + i * 256;
        out[(long)t * cols + c] = (__bf16)(loc[i] * inv * w[c]);
    }
}

// ---------------------------------------------------------------------------
// Generic WMMA GEMM: C[M,N] = A[M,K](bf16 rm) * W[N,K](bf16 rm)^T
// 256 threads = 8 waves; block tile 128x256; wave tile 64x64; K-step 32.
// A tile: per-lane async global->LDS DMA. B tile: single TDM tensor load
// issued by wave 0 (fallback: per-lane async). Double-buffered LDS.
// M multiple of 128; N multiple of 256; K multiple of 32.
// ---------------------------------------------------------------------------
template <typename OutT>
__global__ __launch_bounds__(256) void gemm_bf16_kernel(
    const __bf16* __restrict__ A, const __bf16* __restrict__ W,
    OutT* __restrict__ C, int M, int N, int K)
{
    constexpr int KB = 32;
    constexpr int BN = 256;
    __shared__ __align__(16) __bf16 lA[2][128 * KB];
    __shared__ __align__(16) __bf16 lB[2][BN * KB];

    const int tid  = threadIdx.x;
    const int wave = tid >> 5;
    const int lane = tid & 31;
    const int hl   = lane >> 4;      // half-wave id
    const int ln   = lane & 15;
    const int bm = blockIdx.x * 128;
    const int bn = blockIdx.y * BN;
    const int wm = (wave & 1) * 64;  // wave M offset in block tile
    const int wn = (wave >> 1) * 64; // wave N offset in block tile

    v8f acc[4][4] = {};

    const int ldRow = tid >> 1;          // 0..127
    const int ldCol = (tid & 1) * 16;    // 0 or 16
    const int kTiles = K / KB;

    const unsigned ldsA = lds_addr(&lA[0][0]);
    const unsigned ldsB = lds_addr(&lB[0][0]);
    const unsigned long long baseA = (unsigned long long)A;
#if !HAVE_TDM
    const unsigned long long baseW = (unsigned long long)W;
#endif

    auto loadTile = [&](int kt, int buf) {
        unsigned la = ldsA + (unsigned)(buf * (128 * KB * 2) +
                                        (ldRow * KB + ldCol) * 2);
        int va = (int)(((long)(bm + ldRow) * K + (long)kt * KB + ldCol) * 2);
        ASYNC_B128(la, va, baseA, 0);
        ASYNC_B128(la, va, baseA, 16);
#if HAVE_TDM
        if (wave == 0) {
            tdm_load_tile(ldsB + (unsigned)(buf * (BN * KB * 2)),
                          W + (size_t)bn * K + (size_t)kt * KB,
                          (unsigned)K, (unsigned)(N - bn),
                          (unsigned)KB, (unsigned)BN, (unsigned)K);
        }
#else
        unsigned lb = ldsB + (unsigned)(buf * (BN * KB * 2) + tid * KB * 2);
        int vb = (int)(((long)(bn + tid) * K + (long)kt * KB) * 2);
        ASYNC_B128(lb, vb, baseW, 0);
        ASYNC_B128(lb, vb, baseW, 16);
        ASYNC_B128(lb, vb, baseW, 32);
        ASYNC_B128(lb, vb, baseW, 48);
#endif
    };

    loadTile(0, 0);
    WAIT_ASYNC();
#if HAVE_TDM
    if (wave == 0) wait_tensor0();
#endif
    __syncthreads();

    int buf = 0;
    for (int kt = 0; kt < kTiles; ++kt) {
        if (kt + 1 < kTiles) loadTile(kt + 1, buf ^ 1);
        if (kt + 2 < kTiles) {
            __builtin_prefetch(A + (size_t)(bm + ldRow) * K + (kt + 2) * KB, 0, 0);
        }
        Frag fa[4];
#pragma unroll
        for (int mt = 0; mt < 4; ++mt) {
            const __bf16* base = &lA[buf][(wm + mt * 16 + ln) * KB];
            fa[mt].u[0] = *(const uint4*)(base + hl * 8);
            fa[mt].u[1] = *(const uint4*)(base + 16 + hl * 8);
        }
#pragma unroll
        for (int nt = 0; nt < 4; ++nt) {
            Frag fb;
            const __bf16* base = &lB[buf][(wn + nt * 16 + ln) * KB + hl * 8];
            fb.u[0] = *(const uint4*)(base);
            fb.u[1] = *(const uint4*)(base + 16);
#pragma unroll
            for (int mt = 0; mt < 4; ++mt) {
                acc[mt][nt] = __builtin_amdgcn_wmma_f32_16x16x32_bf16(
                    false, fa[mt].v, false, fb.v, (short)0, acc[mt][nt],
                    false, false);
            }
        }
        WAIT_ASYNC();
#if HAVE_TDM
        if (wave == 0) wait_tensor0();
#endif
        __syncthreads();
        buf ^= 1;
    }

    // Epilogue: C layout: VGPR r -> M = hl*8 + r; lane halves -> N.
#pragma unroll
    for (int mt = 0; mt < 4; ++mt) {
#pragma unroll
        for (int nt = 0; nt < 4; ++nt) {
#pragma unroll
            for (int r = 0; r < 8; ++r) {
                int gr = bm + wm + mt * 16 + hl * 8 + r;
                int gc = bn + wn + nt * 16 + ln;
                C[(size_t)gr * N + gc] = (OutT)acc[mt][nt][r];
            }
        }
    }
}

// ---------------------------------------------------------------------------
// Per-head QK RMSNorm + RoPE + bf16 cache scatter.
// grid.x = T, grid.y = H + HKV + HKV (q heads, k heads, v heads), block = 128.
// ---------------------------------------------------------------------------
__global__ __launch_bounds__(128) void qknorm_rope_kernel(
    const __bf16* __restrict__ qin,   // T x (H*D)
    const __bf16* __restrict__ kin,   // T x (HKV*D)
    const __bf16* __restrict__ vin,   // T x (HKV*D)
    const float* __restrict__ qw, const float* __restrict__ kw,
    const float* __restrict__ cosT, const float* __restrict__ sinT,
    const int* __restrict__ cpos,
    __bf16* __restrict__ qout,        // H x T x D
    __bf16* __restrict__ kcache,      // HKV x S x D
    __bf16* __restrict__ vcache)      // HKV x S x D
{
    const int t = blockIdx.x;
    const int slot = blockIdx.y;
    const int d = threadIdx.x;
    const int cur = cpos[0];
    __shared__ float red[4];

    if (slot >= H + HKV) {            // V: plain copy into cache
        int hk = slot - H - HKV;
        vcache[(size_t)hk * S * D + (size_t)(cur + t) * D + d] =
            vin[(size_t)t * (HKV * D) + hk * D + d];
        return;
    }

    const bool isQ = (slot < H);
    const int head = isQ ? slot : (slot - H);
    const __bf16* src = isQ ? (qin + (size_t)t * (H * D) + head * D)
                            : (kin + (size_t)t * (HKV * D) + head * D);
    const float* w = isQ ? qw : kw;

    float v = (float)src[d];
    float ss = v * v;
#pragma unroll
    for (int off = 16; off >= 1; off >>= 1) ss += __shfl_xor(ss, off, 32);
    if ((threadIdx.x & 31) == 0) red[threadIdx.x >> 5] = ss;
    __syncthreads();
    float tot = red[0] + red[1] + red[2] + red[3];
    float inv = rsqrtf(tot / (float)D + EPS);

    int pd = (d < 64) ? d + 64 : d - 64;
    float y  = v * inv * w[d];
    float y2 = (float)src[pd] * inv * w[pd];
    float rot = (d < 64) ? -y2 : y2;
    float o = y * cosT[(size_t)t * D + d] + rot * sinT[(size_t)t * D + d];

    if (isQ) {
        qout[(size_t)head * T * D + (size_t)t * D + d] = (__bf16)o;
    } else {
        kcache[(size_t)head * S * D + (size_t)(cur + t) * D + d] = (__bf16)o;
    }
}

// ---------------------------------------------------------------------------
// Causal flash attention, bf16 WMMA.
// grid = (T/64, H); block = 128 threads = 4 waves; each wave owns 16 queries.
// Key blocks of 32. K tile via TDM (fallback async DMA); V transposed in LDS.
// ---------------------------------------------------------------------------
__global__ __launch_bounds__(128) void attn_kernel(
    const __bf16* __restrict__ Q,    // H x T x D
    const __bf16* __restrict__ Kc,   // HKV x S x D
    const __bf16* __restrict__ Vc,   // HKV x S x D
    __bf16* __restrict__ Out)        // T x (H*D)
{
    __shared__ __align__(16) __bf16 Kt[32 * 128];
    __shared__ __align__(16) __bf16 Vt[128 * 32];   // [d][s]
    __shared__ __align__(16) __bf16 Pl[4][16 * 32];

    const int h   = blockIdx.y;
    const int hkv = h / NREP;
    const int tid = threadIdx.x;
    const int wave = tid >> 5;
    const int lane = tid & 31;
    const int hl = lane >> 4;
    const int ln = lane & 15;
    const int qRow = blockIdx.x * 64 + wave * 16;

    const __bf16* Qh = Q  + (size_t)h   * T * D;
    const __bf16* Kh = Kc + (size_t)hkv * S * D;
    const __bf16* Vh = Vc + (size_t)hkv * S * D;

    const unsigned ldsK = lds_addr(Kt);
#if !HAVE_TDM
    const unsigned long long baseK = (unsigned long long)Kh;
#endif

    // Q fragments for all 4 K-steps over D=128, held in registers.
    Frag fq[4];
    {
        const __bf16* qr = Qh + (size_t)(qRow + ln) * D;
#pragma unroll
        for (int ks = 0; ks < 4; ++ks) {
            fq[ks].u[0] = *(const uint4*)(qr + ks * 32 + hl * 8);
            fq[ks].u[1] = *(const uint4*)(qr + ks * 32 + 16 + hl * 8);
        }
    }

    float mrow[8], lrow[8];
    v8f accO[8] = {};
#pragma unroll
    for (int r = 0; r < 8; ++r) { mrow[r] = -1e30f; lrow[r] = 0.f; }

    const int nBlocks = blockIdx.x * 2 + 2;  // keys up to block's last query
    for (int b = 0; b < nBlocks; ++b) {
        const int sBase = b * 32;

        // K tile -> LDS via TDM; V tile transposed into LDS via VGPRs.
#if HAVE_TDM
        if (wave == 0) {
            tdm_load_tile(ldsK, Kh + (size_t)sBase * D,
                          (unsigned)D, (unsigned)(S - sBase),
                          (unsigned)D, 32u, (unsigned)D);
        }
#endif
        {
            const int r = tid >> 2, c = (tid & 3) * 32;
#if !HAVE_TDM
            unsigned lk = ldsK + (unsigned)((r * 128 + c) * 2);
            int vk = (int)(((long)(sBase + r) * D + c) * 2);
            ASYNC_B128(lk, vk, baseK, 0);
            ASYNC_B128(lk, vk, baseK, 16);
            ASYNC_B128(lk, vk, baseK, 32);
            ASYNC_B128(lk, vk, baseK, 48);
#endif
            const __bf16* pv = Vh + (size_t)(sBase + r) * D + c;
#pragma unroll
            for (int j = 0; j < 32; ++j) Vt[(c + j) * 32 + r] = pv[j];
        }
        WAIT_ASYNC();
#if HAVE_TDM
        if (wave == 0) wait_tensor0();
#endif
        __syncthreads();

        if (sBase <= qRow + 15) {   // wave-uniform causal skip
            // scores: S = Q K^T, two 16-key tiles
            v8f sc[2] = {};
#pragma unroll
            for (int ct = 0; ct < 2; ++ct) {
#pragma unroll
                for (int ks = 0; ks < 4; ++ks) {
                    Frag fb;
                    const __bf16* kb =
                        &Kt[(ct * 16 + ln) * 128 + ks * 32 + hl * 8];
                    fb.u[0] = *(const uint4*)(kb);
                    fb.u[1] = *(const uint4*)(kb + 16);
                    sc[ct] = __builtin_amdgcn_wmma_f32_16x16x32_bf16(
                        false, fq[ks].v, false, fb.v, (short)0, sc[ct],
                        false, false);
                }
            }
            // scale + causal mask + online softmax
#pragma unroll
            for (int r = 0; r < 8; ++r) {
                const int qg = qRow + hl * 8 + r;
                float a0 = sc[0][r] * SCALE;
                float a1 = sc[1][r] * SCALE;
                if (sBase + ln      > qg) a0 = -1e30f;
                if (sBase + 16 + ln > qg) a1 = -1e30f;
                float mx = fmaxf(a0, a1);
#pragma unroll
                for (int off = 8; off >= 1; off >>= 1)
                    mx = fmaxf(mx, __shfl_xor(mx, off, 32));
                float mnew  = fmaxf(mrow[r], mx);
                float alpha = __expf(mrow[r] - mnew);
                float p0 = __expf(a0 - mnew);
                float p1 = __expf(a1 - mnew);
                float ps = p0 + p1;
#pragma unroll
                for (int off = 8; off >= 1; off >>= 1)
                    ps += __shfl_xor(ps, off, 32);
                lrow[r] = lrow[r] * alpha + ps;
                mrow[r] = mnew;
#pragma unroll
                for (int nt = 0; nt < 8; ++nt) accO[nt][r] *= alpha;
                Pl[wave][(hl * 8 + r) * 32 + ln]      = (__bf16)p0;
                Pl[wave][(hl * 8 + r) * 32 + 16 + ln] = (__bf16)p1;
            }
            // P (16x32, C-layout in LDS) -> A fragment
            Frag fp;
            {
                const __bf16* pb = &Pl[wave][ln * 32];
                fp.u[0] = *(const uint4*)(pb + hl * 8);
                fp.u[1] = *(const uint4*)(pb + 16 + hl * 8);
            }
            // O += P V
#pragma unroll
            for (int nt = 0; nt < 8; ++nt) {
                Frag fv;
                const __bf16* vb = &Vt[(nt * 16 + ln) * 32 + hl * 8];
                fv.u[0] = *(const uint4*)(vb);
                fv.u[1] = *(const uint4*)(vb + 16);
                accO[nt] = __builtin_amdgcn_wmma_f32_16x16x32_bf16(
                    false, fp.v, false, fv.v, (short)0, accO[nt],
                    false, false);
            }
        }
        __syncthreads();
    }

    // Normalize and store to (T, H*D)
#pragma unroll
    for (int r = 0; r < 8; ++r) {
        const float inv = 1.f / lrow[r];
        const int t = qRow + hl * 8 + r;
#pragma unroll
        for (int nt = 0; nt < 8; ++nt) {
            Out[(size_t)t * (H * D) + h * D + nt * 16 + ln] =
                (__bf16)(accO[nt][r] * inv);
        }
    }
}

// ---------------------------------------------------------------------------
// Elementwise epilogues
// ---------------------------------------------------------------------------
__global__ __launch_bounds__(256) void residual_add_kernel(
    const float* __restrict__ conv,   // HID x T (transposed layout)
    const float* __restrict__ oproj,  // T x HID
    float* __restrict__ hid)          // T x HID
{
    long i = (long)blockIdx.x * blockDim.x + threadIdx.x;
    long n = (long)T * HID;
    long stride = (long)gridDim.x * blockDim.x;
    for (; i < n; i += stride) {
        long t = i / HID, c = i % HID;
        hid[i] = conv[c * (long)T + t] + oproj[i];
    }
}

__global__ __launch_bounds__(256) void silu_mul_kernel(
    const __bf16* __restrict__ gu,    // T x (2*IDIM)
    __bf16* __restrict__ act)         // T x IDIM
{
    long i = (long)blockIdx.x * blockDim.x + threadIdx.x;
    long n = (long)T * IDIM;
    long stride = (long)gridDim.x * blockDim.x;
    for (; i < n; i += stride) {
        long t = i / IDIM, c = i % IDIM;
        float g = (float)gu[t * (2 * (long)IDIM) + c];
        float u = (float)gu[t * (2 * (long)IDIM) + IDIM + c];
        act[i] = (__bf16)((g / (1.f + __expf(-g))) * u);
    }
}

__global__ __launch_bounds__(256) void final_out_kernel(
    const float* __restrict__ hid,    // T x HID
    const float* __restrict__ down,   // T x HID
    float* __restrict__ out)          // HID x T (transposed output)
{
    long i = (long)blockIdx.x * blockDim.x + threadIdx.x;
    long n = (long)T * HID;
    long stride = (long)gridDim.x * blockDim.x;
    for (; i < n; i += stride) {
        long t = i / HID, c = i % HID;
        out[c * (long)T + t] = hid[i] + down[i];
    }
}

// ---------------------------------------------------------------------------
// Host launcher
// ---------------------------------------------------------------------------
extern "C" void kernel_launch(void* const* d_in, const int* in_sizes, int n_in,
                              void* d_out, int out_size, void* d_ws,
                              size_t ws_size, hipStream_t stream)
{
    const float* hidden_conv = (const float*)d_in[0];
    const float* cosT        = (const float*)d_in[1];
    const float* sinT        = (const float*)d_in[2];
    // d_in[3] causal_mask, d_in[4] kv_cache: handled analytically / zeros
    const float* wq  = (const float*)d_in[5];
    const float* wk  = (const float*)d_in[6];
    const float* wv  = (const float*)d_in[7];
    const float* wo  = (const float*)d_in[8];
    const float* wgu = (const float*)d_in[9];
    const float* wdn = (const float*)d_in[10];
    const float* iln = (const float*)d_in[11];
    const float* pln = (const float*)d_in[12];
    const float* qnw = (const float*)d_in[13];
    const float* knw = (const float*)d_in[14];
    const int*   cpos = (const int*)d_in[15];
    float* out = (float*)d_out;

    char* ws = (char*)d_ws;
    size_t off = 0;
    auto alloc = [&](size_t bytes) -> char* {
        char* p = ws + off;
        off += (bytes + 255) & ~(size_t)255;
        return p;
    };

    __bf16* wq_bf  = (__bf16*)alloc((size_t)HID * HID * 2);
    __bf16* wk_bf  = (__bf16*)alloc((size_t)HKV * D * HID * 2);
    __bf16* wv_bf  = (__bf16*)alloc((size_t)HKV * D * HID * 2);
    __bf16* wo_bf  = (__bf16*)alloc((size_t)HID * H * D * 2);
    __bf16* wgu_bf = (__bf16*)alloc((size_t)2 * IDIM * HID * 2);
    __bf16* wdn_bf = (__bf16*)alloc((size_t)HID * IDIM * 2);
    __bf16* h_bf   = (__bf16*)alloc((size_t)T * HID * 2);
    __bf16* q_bf   = (__bf16*)alloc((size_t)T * H * D * 2);
    __bf16* k_bf   = (__bf16*)alloc((size_t)T * HKV * D * 2);
    __bf16* v_bf   = (__bf16*)alloc((size_t)T * HKV * D * 2);
    __bf16* qr_bf  = (__bf16*)alloc((size_t)H * T * D * 2);
    __bf16* kc_bf  = (__bf16*)alloc((size_t)HKV * S * D * 2);
    __bf16* vc_bf  = (__bf16*)alloc((size_t)HKV * S * D * 2);
    __bf16* ao_bf  = (__bf16*)alloc((size_t)T * H * D * 2);
    float*  op_f   = (float*) alloc((size_t)T * HID * 4);
    float*  hid_f  = (float*) alloc((size_t)T * HID * 4);
    __bf16* h2_bf  = (__bf16*)alloc((size_t)T * HID * 2);
    __bf16* gu_bf  = (__bf16*)alloc((size_t)T * 2 * IDIM * 2);
    __bf16* act_bf = (__bf16*)alloc((size_t)T * IDIM * 2);
    float*  dn_f   = (float*) alloc((size_t)T * HID * 4);

    // 1) weights -> bf16
    auto cvt = [&](const float* src, __bf16* dst, long n) {
        f32_to_bf16_kernel<<<512, 256, 0, stream>>>(src, dst, n);
    };
    cvt(wq,  wq_bf,  (long)HID * HID);
    cvt(wk,  wk_bf,  (long)HKV * D * HID);
    cvt(wv,  wv_bf,  (long)HKV * D * HID);
    cvt(wo,  wo_bf,  (long)HID * H * D);
    cvt(wgu, wgu_bf, (long)2 * IDIM * HID);
    cvt(wdn, wdn_bf, (long)HID * IDIM);

    // 2) input RMSNorm (reads transposed hidden_conv)
    rms_kernel<<<T, 256, 0, stream>>>(hidden_conv, iln, h_bf, HID, 1L, (long)T);

    // 3) QKV projections (block tile 128x256)
    gemm_bf16_kernel<__bf16><<<dim3(T / 128, (H * D) / 256), 256, 0, stream>>>(
        h_bf, wq_bf, q_bf, T, H * D, HID);
    gemm_bf16_kernel<__bf16><<<dim3(T / 128, (HKV * D) / 256), 256, 0, stream>>>(
        h_bf, wk_bf, k_bf, T, HKV * D, HID);
    gemm_bf16_kernel<__bf16><<<dim3(T / 128, (HKV * D) / 256), 256, 0, stream>>>(
        h_bf, wv_bf, v_bf, T, HKV * D, HID);

    // 4) QK norm + RoPE + cache
    qknorm_rope_kernel<<<dim3(T, H + 2 * HKV), 128, 0, stream>>>(
        q_bf, k_bf, v_bf, qnw, knw, cosT, sinT, cpos, qr_bf, kc_bf, vc_bf);

    // 5) causal flash attention
    attn_kernel<<<dim3(T / 64, H), 128, 0, stream>>>(qr_bf, kc_bf, vc_bf, ao_bf);

    // 6) O projection + residual
    gemm_bf16_kernel<float><<<dim3(T / 128, HID / 256), 256, 0, stream>>>(
        ao_bf, wo_bf, op_f, T, HID, H * D);
    residual_add_kernel<<<1024, 256, 0, stream>>>(hidden_conv, op_f, hid_f);

    // 7) post RMSNorm (row-major hidden)
    rms_kernel<<<T, 256, 0, stream>>>(hid_f, pln, h2_bf, HID, (long)HID, 1L);

    // 8) MLP
    gemm_bf16_kernel<__bf16><<<dim3(T / 128, (2 * IDIM) / 256), 256, 0, stream>>>(
        h2_bf, wgu_bf, gu_bf, T, 2 * IDIM, HID);
    silu_mul_kernel<<<1024, 256, 0, stream>>>(gu_bf, act_bf);
    gemm_bf16_kernel<float><<<dim3(T / 128, HID / 256), 256, 0, stream>>>(
        act_bf, wdn_bf, dn_f, T, HID, IDIM);

    // 9) final residual + transpose to (1, HID, 1, T)
    final_out_kernel<<<1024, 256, 0, stream>>>(hid_f, dn_f, out);

    (void)in_sizes; (void)n_in; (void)out_size; (void)ws_size;
}
